// MyModel_2_84189948936658
// MI455X (gfx1250) — matrix operand order repaired and checked by reading, not compiled, not run on hardware
//
#include <hip/hip_runtime.h>
#include <hip/hip_bf16.h>
#include <math.h>

// ---------------------------------------------------------------------------
// Types
// ---------------------------------------------------------------------------
typedef __bf16 bf16_t;
typedef __attribute__((ext_vector_type(16))) __bf16 v16bf;
typedef __attribute__((ext_vector_type(8)))  __bf16 v8bf;
typedef __attribute__((ext_vector_type(8)))  float  v8f;
typedef __attribute__((ext_vector_type(4)))  unsigned v4u;
typedef __attribute__((ext_vector_type(8)))  int      v8i_t;
typedef __attribute__((ext_vector_type(4)))  int      v4i_t;

#if __has_builtin(__builtin_amdgcn_tensor_load_to_lds)
#define USE_TDM 1
#else
#define USE_TDM 0
#endif

// ---------------------------------------------------------------------------
// Problem constants
// ---------------------------------------------------------------------------
#define B_   1024
#define T_   20
#define E_   1024
#define H_   1024
#define H2_  2048
#define DT_  50
#define NP_  64           // classifier N padded to WMMA multiple
#define BT_  (B_ * T_)

// GEMM tile config: 4 waves (128 threads), each wave computes 32x32,
// BK=64 -> 8 WMMAs per staged tile, double-buffered in LDS.
#define BM 64
#define BN 64
#define BK 64

// ---------------------------------------------------------------------------
// fp32 -> bf16 conversion (plain)
// ---------------------------------------------------------------------------
__global__ __launch_bounds__(256) void k_cvt_bf16(const float* __restrict__ in,
                                                  bf16_t* __restrict__ out, int n) {
  int i = blockIdx.x * 256 + threadIdx.x;
  if (i < n) out[i] = (bf16_t)in[i];
}

// fp32 [K][N] -> bf16 [Npad][K] (transposed, zero-padded in N)
__global__ __launch_bounds__(256) void k_cvt_T(const float* __restrict__ in,
                                               bf16_t* __restrict__ out,
                                               int K, int N, int Npad) {
  int i = blockIdx.x * 256 + threadIdx.x;
  if (i >= Npad * K) return;
  int n = i / K, k = i % K;
  float v = (n < N) ? in[(size_t)k * N + n] : 0.0f;
  out[(size_t)n * K + k] = (bf16_t)v;
}

__global__ void k_pad_bias(const float* __restrict__ in, float* __restrict__ out,
                           int N, int Npad) {
  int i = threadIdx.x;
  if (i < Npad) out[i] = (i < N) ? in[i] : 0.0f;
}

// ---------------------------------------------------------------------------
// TDM 2D tile load: global [tile_h x tile_w] (elements, row stride = stride
// elements) -> contiguous LDS at lds_off. bf16 elements (data_size=1 -> 2B).
// D# layout per CDNA5 ISA.
// All arguments must be wave-uniform; EXEC is ignored by TDM.
// This toolchain exposes the 6-arg builtin:
//   (uint32x4 g0, int32x8 g1, int32x4 g2, int32x4 g3, int32x8 extra, i32 cpol)
// Groups 2/3 (+extra) are zero for <=2D tensors (VADDR2/3 disabled).
// ---------------------------------------------------------------------------
#if USE_TDM
__device__ __forceinline__ void tdm_load_2d(unsigned lds_off, const void* gaddr,
                                            int tile_w, int tile_h,
                                            unsigned long long stride) {
  unsigned long long ga = (unsigned long long)gaddr;
  v4u g0;
  g0[0] = 1u;                                            // count=1, user desc
  g0[1] = lds_off;                                       // lds_addr
  g0[2] = (unsigned)(ga & 0xFFFFFFFFull);                // global_addr[31:0]
  g0[3] = (unsigned)((ga >> 32) & 0x01FFFFFFull)         // global_addr[56:32]
        | (2u << 30);                                    // type=2 ("image")
  const unsigned td0 = (unsigned)stride;                 // tensor_dim0 (elems)
  const unsigned td1 = 1u << 20;                         // tensor_dim1 (large)
  v8i_t g1;
  g1[0] = (int)(1u << 16);                               // data_size=1 (2B)
  g1[1] = (int)((td0 & 0xFFFFu) << 16);                  // tensor_dim0 lo16
  g1[2] = (int)((td0 >> 16) | ((td1 & 0xFFFFu) << 16));  // dim0 hi16 | dim1 lo16
  g1[3] = (int)((td1 >> 16) | ((unsigned)tile_w << 16)); // dim1 hi16 | tile_dim0
  g1[4] = (int)(unsigned)tile_h;                         // tile_dim1 (tile_dim2=0)
  g1[5] = (int)(unsigned)(stride & 0xFFFFFFFFull);       // dim0_stride[31:0]
  g1[6] = (int)(unsigned)((stride >> 32) & 0xFFFFull);   // dim0_stride[47:32]
  g1[7] = 0;
  v4i_t z4 = {};
  v8i_t z8 = {};
  __builtin_amdgcn_tensor_load_to_lds(g0, g1, z4, z4, z8, 0);
}
#endif

// ---------------------------------------------------------------------------
// Generic bf16 WMMA GEMM:  C = act( A1@W1 + A2@W2 + Cin + bias )
//   A*  : bf16 row-major [M][K*] with leading dim lda*
//   W*  : bf16 N-major  [N][K*] (pre-transposed), leading dim = K*
//   Cin : optional bf16 addend, leading dim ldcin
//   bias: optional fp32 [N];  act: 0=none, 1=tanh
// Requires M%64==0, N%64==0, K%64==0 (true for all call sites).
// ---------------------------------------------------------------------------
__global__ __launch_bounds__(128) void k_gemm(
    const bf16_t* __restrict__ A1, int lda1, const bf16_t* __restrict__ W1, int K1,
    const bf16_t* __restrict__ A2, int lda2, const bf16_t* __restrict__ W2, int K2,
    const float* __restrict__ bias, const bf16_t* __restrict__ Cin, int ldcin,
    int act,
    bf16_t* __restrict__ outb, int ldob, float* __restrict__ outf, int ldof)
{
  __shared__ bf16_t sA[2][BM][BK];   // 16 KB
  __shared__ bf16_t sB[2][BN][BK];   // 16 KB

  const int tid  = threadIdx.x;
  const int lane = tid & 31;
  const int wv   = tid >> 5;   // wave 0..3
  const int wm   = wv >> 1;    // 0..1  (M sub-block)
  const int wn   = wv & 1;     // 0..1  (N sub-block)
  const int row0 = blockIdx.y * BM;
  const int col0 = blockIdx.x * BN;

  v8f zero = {};
  v8f acc[2][2];
  acc[0][0] = zero; acc[0][1] = zero; acc[1][0] = zero; acc[1][1] = zero;

  const int n1 = A1 ? (K1 / BK) : 0;
  const int n2 = A2 ? (K2 / BK) : 0;
  const int ns = n1 + n2;

  // Resolve pipeline step s -> (A, W, lda, K, k0)
  auto stage_src = [&](int s, const bf16_t*& A, const bf16_t*& W,
                       int& lda, int& K, int& k0) {
    if (s < n1) { A = A1; W = W1; lda = lda1; K = K1; k0 = s * BK; }
    else        { A = A2; W = W2; lda = lda2; K = K2; k0 = (s - n1) * BK; }
  };

  auto issue = [&](int s, int buf) {
    const bf16_t* A; const bf16_t* W; int lda, K, k0;
    stage_src(s, A, W, lda, K, k0);
#if USE_TDM
    tdm_load_2d((unsigned)(size_t)&sA[buf][0][0],
                &A[(size_t)row0 * lda + k0], BK, BM, (unsigned long long)lda);
    tdm_load_2d((unsigned)(size_t)&sB[buf][0][0],
                &W[(size_t)col0 * K + k0], BK, BN, (unsigned long long)K);
#endif
  };

#if !USE_TDM
  // Fallback: statically unrolled cooperative staging (no waterfall loops).
  auto stage_direct = [&](int s, int buf) {
    const bf16_t* A; const bf16_t* W; int lda, K, k0;
    stage_src(s, A, W, lda, K, k0);
    const int r = tid >> 1;            // 0..63
    const int c = (tid & 1) * 32;      // 0 or 32
    #pragma unroll
    for (int j = 0; j < 4; ++j)
      *(v8bf*)&sA[buf][r][c + 8 * j] =
          *(const v8bf*)&A[(size_t)(row0 + r) * lda + k0 + c + 8 * j];
    #pragma unroll
    for (int j = 0; j < 4; ++j)
      *(v8bf*)&sB[buf][r][c + 8 * j] =
          *(const v8bf*)&W[(size_t)(col0 + r) * K + k0 + c + 8 * j];
  };
#endif

  if (ns > 0) {
#if USE_TDM
    if (wv == 0) {                      // one wave drives the TDM
      issue(0, 0);
      __builtin_amdgcn_s_wait_tensorcnt(0);
    }
    __syncthreads();
#else
    stage_direct(0, 0);
    __syncthreads();
#endif
    for (int s = 0; s < ns; ++s) {
      const int cur = s & 1;
#if USE_TDM
      if (wv == 0 && s + 1 < ns) issue(s + 1, cur ^ 1);  // overlap DMA w/ math
#else
      if (s + 1 < ns) stage_direct(s + 1, cur ^ 1);
#endif
      // ---- 8 WMMAs from the current LDS buffers ----
      #pragma unroll
      for (int kk = 0; kk < 2; ++kk) {
        // A frag (16x32): lane l -> row l%16; lanes 0-15: K {0..15},
        // lanes 16-31: K {16..31} within this 32-wide k-step.
        v16bf af[2];
        #pragma unroll
        for (int mi = 0; mi < 2; ++mi) {
          int r  = wm * 32 + mi * 16 + (lane & 15);
          int ko = kk * 32 + (lane >> 4) * 16;
          v8bf lo = *(const v8bf*)&sA[cur][r][ko];
          v8bf hi = *(const v8bf*)&sA[cur][r][ko + 8];
          #pragma unroll
          for (int j = 0; j < 8; ++j) { af[mi][j] = lo[j]; af[mi][j + 8] = hi[j]; }
        }
        // B frag (32x16, N-major): lane l -> col l%16; contiguous 16 K elems.
        v16bf bfr[2];
        #pragma unroll
        for (int ni = 0; ni < 2; ++ni) {
          int c  = wn * 32 + ni * 16 + (lane & 15);
          int kb = kk * 32 + (lane >> 4) * 16;
          bfr[ni] = *(const v16bf*)&sB[cur][c][kb];
        }
        #pragma unroll
        for (int mi = 0; mi < 2; ++mi)
          #pragma unroll
          for (int ni = 0; ni < 2; ++ni)
            acc[mi][ni] = __builtin_amdgcn_wmma_f32_16x16x32_bf16(
                false, af[mi], false, bfr[ni], (short)0, acc[mi][ni], false, false);
      }
      __syncthreads();                       // everyone done reading buf `cur`
#if USE_TDM
      if (wv == 0 && s + 1 < ns) __builtin_amdgcn_s_wait_tensorcnt(0);
#endif
      __syncthreads();                       // next buffer published
    }
  }

  // Epilogue. C/D layout: VGPR v, lane l -> M = v + (l>=16 ? 8 : 0), N = l%16.
  const int ln16 = lane & 15;
  const int mhi  = (lane >> 4) * 8;
  #pragma unroll
  for (int mi = 0; mi < 2; ++mi) {
    #pragma unroll
    for (int ni = 0; ni < 2; ++ni) {
      #pragma unroll
      for (int v = 0; v < 8; ++v) {
        int gr = row0 + wm * 32 + mi * 16 + mhi + v;
        int gc = col0 + wn * 32 + ni * 16 + ln16;
        float x = acc[mi][ni][v];
        if (Cin)  x += (float)Cin[(size_t)gr * ldcin + gc];
        if (bias) x += bias[gc];
        if (act)  x = tanhf(x);
        if (outb) outb[(size_t)gr * ldob + gc] = (bf16_t)x;
        if (outf) outf[(size_t)gr * ldof + gc] = x;
      }
    }
  }
}

// ---------------------------------------------------------------------------
// LayerNorm over H=1024 (block = one row of B). Optionally also writes the
// result into the [B][T][2H] concat buffer at (possibly scattered) position.
// ---------------------------------------------------------------------------
__global__ __launch_bounds__(256) void k_ln(
    const bf16_t* __restrict__ h, const float* __restrict__ gw,
    const float* __restrict__ gb, bf16_t* __restrict__ lnout,
    bf16_t* __restrict__ outbuf, const int* __restrict__ olen,
    int t, int coloff)
{
  const int b = blockIdx.x, tid = threadIdx.x;
  const bf16_t* row = h + (size_t)b * H_;
  float vals[4], s = 0.f, s2 = 0.f;
  #pragma unroll
  for (int i = 0; i < 4; ++i) {
    float v = (float)row[tid + i * 256];
    vals[i] = v; s += v; s2 += v * v;
  }
  __shared__ float r1[256], r2[256];
  r1[tid] = s; r2[tid] = s2;
  __syncthreads();
  for (int off = 128; off > 0; off >>= 1) {
    if (tid < off) { r1[tid] += r1[tid + off]; r2[tid] += r2[tid + off]; }
    __syncthreads();
  }
  const float mu   = r1[0] * (1.0f / H_);
  const float var  = r2[0] * (1.0f / H_) - mu * mu;
  const float rstd = rsqrtf(var + 1e-5f);

  int pos = t;
  if (olen) {  // r2l scatter: pos = (T - min(len,T) - t) mod T
    int L = olen[b]; if (L > T_) L = T_;
    pos = ((T_ - L - t) % T_ + T_) % T_;
  }
  #pragma unroll
  for (int i = 0; i < 4; ++i) {
    int idx = tid + i * 256;
    float y = (vals[i] - mu) * rstd * gw[idx] + gb[idx];
    lnout[(size_t)b * H_ + idx] = (bf16_t)y;
    if (outbuf)
      outbuf[((size_t)b * T_ + pos) * H2_ + coloff + idx] = (bf16_t)y;
  }
}

// ---------------------------------------------------------------------------
// Softmax over the TIME axis: one thread per (b, d) pair.
// logits are [B][T][NP_] fp32; output [B][T][DT_] fp32.
// ---------------------------------------------------------------------------
__global__ __launch_bounds__(256) void k_softmax_t(const float* __restrict__ lg,
                                                   float* __restrict__ out) {
  int i = blockIdx.x * 256 + threadIdx.x;
  if (i >= B_ * DT_) return;
  int b = i / DT_, d = i % DT_;
  const float* base = lg + (size_t)b * T_ * NP_ + d;
  float mx = -3.4e38f;
  #pragma unroll
  for (int t = 0; t < T_; ++t) mx = fmaxf(mx, base[t * NP_]);
  float e[T_], s = 0.f;
  #pragma unroll
  for (int t = 0; t < T_; ++t) { e[t] = __expf(base[t * NP_] - mx); s += e[t]; }
  float inv = 1.f / s;
  #pragma unroll
  for (int t = 0; t < T_; ++t)
    out[((size_t)b * T_ + t) * DT_ + d] = e[t] * inv;
}

// ---------------------------------------------------------------------------
// Host orchestration
// ---------------------------------------------------------------------------
extern "C" void kernel_launch(void* const* d_in, const int* in_sizes, int n_in,
                              void* d_out, int out_size, void* d_ws, size_t ws_size,
                              hipStream_t stream) {
  (void)in_sizes; (void)n_in; (void)out_size; (void)ws_size;
  const float* xin[2]  = {(const float*)d_in[0], (const float*)d_in[1]};
  const int*   olen    = (const int*)d_in[2];
  const float* embf[2] = {(const float*)d_in[3], (const float*)d_in[4]};
  const float* wxf[2]  = {(const float*)d_in[5], (const float*)d_in[8]};
  const float* whf[2]  = {(const float*)d_in[6], (const float*)d_in[9]};
  const float* bptr[2] = {(const float*)d_in[7], (const float*)d_in[10]};
  const float* lnw     = (const float*)d_in[11];
  const float* lnb     = (const float*)d_in[12];
  const float* clfw    = (const float*)d_in[13];
  const float* clfb    = (const float*)d_in[14];
  float* out = (float*)d_out;

  // ---- workspace carve-up (base is 256B aligned) ----
  char* base = (char*)d_ws;
  size_t off = 0;
  auto alloc = [&](size_t bytes) -> char* {
    char* p = base + off;
    off += (bytes + 255) & ~(size_t)255;
    return p;
  };
  bf16_t* Xb[2];
  Xb[0] = (bf16_t*)alloc((size_t)BT_ * E_ * 2);
  Xb[1] = (bf16_t*)alloc((size_t)BT_ * E_ * 2);
  bf16_t* OUT = Xb[0];  // alias: 80MB region reused for [B][T][2H] after embedding
  bf16_t *embT[2], *wxT[2][3], *whT[2][3], *XE[2], *XW0[2], *hb[2][3][2], *lnbuf[2];
  for (int d = 0; d < 2; ++d) embT[d] = (bf16_t*)alloc((size_t)H_ * E_ * 2);
  for (int d = 0; d < 2; ++d)
    for (int l = 0; l < 3; ++l) wxT[d][l] = (bf16_t*)alloc((size_t)H_ * H_ * 2);
  for (int d = 0; d < 2; ++d)
    for (int l = 0; l < 3; ++l) whT[d][l] = (bf16_t*)alloc((size_t)H_ * H_ * 2);
  bf16_t* clfT  = (bf16_t*)alloc((size_t)NP_ * H2_ * 2);
  float*  clfbp = (float*)alloc((size_t)NP_ * 4);
  for (int d = 0; d < 2; ++d) XE[d]  = (bf16_t*)alloc((size_t)BT_ * H_ * 2);
  for (int d = 0; d < 2; ++d) XW0[d] = (bf16_t*)alloc((size_t)BT_ * H_ * 2);
  for (int d = 0; d < 2; ++d)
    for (int l = 0; l < 3; ++l)
      for (int pp = 0; pp < 2; ++pp) hb[d][l][pp] = (bf16_t*)alloc((size_t)B_ * H_ * 2);
  for (int d = 0; d < 2; ++d) lnbuf[d] = (bf16_t*)alloc((size_t)B_ * H_ * 2);
  float* logits = (float*)alloc((size_t)BT_ * NP_ * 4);

  // ---- precision conversion / weight transposition ----
  const int nX = BT_ * E_;
  for (int d = 0; d < 2; ++d) {
    k_cvt_bf16<<<(nX + 255) / 256, 256, 0, stream>>>(xin[d], Xb[d], nX);
    k_cvt_T<<<((H_ * E_) + 255) / 256, 256, 0, stream>>>(embf[d], embT[d], E_, H_, H_);
    for (int l = 0; l < 3; ++l) {
      k_cvt_T<<<((H_ * H_) + 255) / 256, 256, 0, stream>>>(
          wxf[d] + (size_t)l * H_ * H_, wxT[d][l], H_, H_, H_);
      k_cvt_T<<<((H_ * H_) + 255) / 256, 256, 0, stream>>>(
          whf[d] + (size_t)l * H_ * H_, whT[d][l], H_, H_, H_);
    }
  }
  k_cvt_T<<<((NP_ * H2_) + 255) / 256, 256, 0, stream>>>(clfw, clfT, H2_, DT_, NP_);
  k_pad_bias<<<1, 64, 0, stream>>>(clfb, clfbp, DT_, NP_);

  auto gemm = [&](int M, int N,
                  const bf16_t* A1, int lda1, const bf16_t* W1, int K1,
                  const bf16_t* A2, int lda2, const bf16_t* W2, int K2,
                  const float* bias, const bf16_t* Cin, int ldcin, int act,
                  bf16_t* ob, int ldob, float* of, int ldof) {
    dim3 g(N / BN, M / BM);
    k_gemm<<<g, 128, 0, stream>>>(A1, lda1, W1, K1, A2, lda2, W2, K2,
                                  bias, Cin, ldcin, act, ob, ldob, of, ldof);
  };

  // ---- hoisted time-independent GEMMs ----
  // XE = X @ emb ;  XW0 = XE @ Wx0 + b0   (consumed as the layer-0 Cin per t)
  for (int d = 0; d < 2; ++d) {
    gemm(BT_, H_, Xb[d], E_, embT[d], E_, nullptr, 0, nullptr, 0,
         nullptr, nullptr, 0, 0, XE[d], H_, nullptr, 0);
    gemm(BT_, H_, XE[d], H_, wxT[d][0], H_, nullptr, 0, nullptr, 0,
         bptr[d], nullptr, 0, 0, XW0[d], H_, nullptr, 0);
  }

  // ---- sequential RNN over T; both directions + fused x/h GEMMs ----
  for (int t = 0; t < T_; ++t) {
    int w = t & 1, r = w ^ 1;
    for (int d = 0; d < 2; ++d) {
      const int* ol = (d == 1) ? olen : nullptr;
      // layer 0: h = tanh(hprev@Wh0 + XW0[:,t,:])   (h0 == 0 -> skip GEMM at t=0)
      gemm(B_, H_, (t ? hb[d][0][r] : nullptr), H_, whT[d][0], H_,
           nullptr, 0, nullptr, 0,
           nullptr, XW0[d] + (size_t)t * H_, T_ * H_, 1,
           hb[d][0][w], H_, nullptr, 0);
      k_ln<<<B_, 256, 0, stream>>>(hb[d][0][w], lnw, lnb, lnbuf[d],
                                   nullptr, nullptr, t, 0);
      // layer 1: h = tanh(ln0@Wx1 + hprev@Wh1 + b1)
      gemm(B_, H_, lnbuf[d], H_, wxT[d][1], H_,
           (t ? hb[d][1][r] : nullptr), H_, whT[d][1], H_,
           bptr[d] + H_, nullptr, 0, 1, hb[d][1][w], H_, nullptr, 0);
      k_ln<<<B_, 256, 0, stream>>>(hb[d][1][w], lnw, lnb, lnbuf[d],
                                   nullptr, nullptr, t, 0);
      // layer 2: h = tanh(ln1@Wx2 + hprev@Wh2 + b2); LN writes concat buffer
      gemm(B_, H_, lnbuf[d], H_, wxT[d][2], H_,
           (t ? hb[d][2][r] : nullptr), H_, whT[d][2], H_,
           bptr[d] + 2 * H_, nullptr, 0, 1, hb[d][2][w], H_, nullptr, 0);
      k_ln<<<B_, 256, 0, stream>>>(hb[d][2][w], lnw, lnb, lnbuf[d],
                                   OUT, ol, t, d * H_);
    }
  }

  // ---- classifier + softmax over time ----
  gemm(BT_, NP_, OUT, H2_, clfT, H2_, nullptr, 0, nullptr, 0,
       clfbp, nullptr, 0, 0, nullptr, 0, logits, NP_);
  k_softmax_t<<<(B_ * DT_ + 255) / 256, 256, 0, stream>>>(logits, out);
}
